// LSTM_CRF_20727512171134
// MI455X (gfx1250) — matrix operand order repaired and checked by reading, not compiled, not run on hardware
//
#include <hip/hip_runtime.h>
#include <hip/hip_bf16.h>

// ---------------- problem constants ----------------
#define BB 64
#define TT 512
#define DD 768
#define HH 512
#define GG 2048   // 4*H
#define NTAGS 16
#define NWG_LSTM 32   // 16 per direction, persistent, globally synced

// ---------------- bf16 helpers ----------------
__device__ __forceinline__ unsigned short f2bf(float f) {
  unsigned u = __float_as_uint(f);
  u += 0x7FFFu + ((u >> 16) & 1u);          // round-to-nearest-even
  return (unsigned short)(u >> 16);
}
__device__ __forceinline__ float bf2f(unsigned short h) {
  return __uint_as_float(((unsigned)h) << 16);
}
__device__ __forceinline__ float sigf(float x) { return 1.0f / (1.0f + __expf(-x)); }

// ---------------- WMMA types (CDNA5 gfx1250) ----------------
typedef __attribute__((ext_vector_type(16))) __bf16 v16bf;
typedef __attribute__((ext_vector_type(8)))  float  v8f;
union Frag16 { uint4 u[2]; v16bf v; };

// ---------------- utility kernels ----------------
__global__ void cvt_f32_bf16(const float* __restrict__ src,
                             unsigned short* __restrict__ dst, int n) {
  int i = blockIdx.x * blockDim.x + threadIdx.x;
  int stride = gridDim.x * blockDim.x;
  for (; i < n; i += stride) dst[i] = f2bf(src[i]);
}

__global__ void zero_u16(unsigned short* p, int n) {
  int i = blockIdx.x * blockDim.x + threadIdx.x;
  int stride = gridDim.x * blockDim.x;
  for (; i < n; i += stride) p[i] = 0;
}

__global__ void zero_u32(unsigned int* p, int n) {
  int i = blockIdx.x * blockDim.x + threadIdx.x;
  if (i < n) p[i] = 0;
}

// ---------------- persistent BiLSTM recurrence ----------------
// grid = 32 blocks x 256 threads. Block bx: dir = bx>>4, owns h-cols
// [ (bx&15)*32 , +32 ).  Per step: gates(64x128) = X_t @ Wih^T + h @ Whh^T
// via v_wmma_f32_16x16x32_bf16 (double-buffered frags), then LSTM cell update.
__global__ void __launch_bounds__(256)
lstm_kernel(const unsigned short* __restrict__ xbf,     // (B,T,D) bf16
            const unsigned short* __restrict__ wih_f,   // (4H,D) bf16
            const unsigned short* __restrict__ whh_f,   // (4H,H) bf16
            const unsigned short* __restrict__ wih_b,
            const unsigned short* __restrict__ whh_b,
            const float* __restrict__ bih_f, const float* __restrict__ bhh_f,
            const float* __restrict__ bih_b, const float* __restrict__ bhh_b,
            const int* __restrict__ lengths,
            unsigned short* __restrict__ hbuf,          // 4 x (B*H) bf16 ping-pong
            unsigned short* __restrict__ hseq_f,        // (B,T,H) bf16
            unsigned short* __restrict__ hseq_b,
            unsigned int* __restrict__ cnt)             // global barrier counter
{
  __shared__ float gates[64 * 128];   // 32 KB
  __shared__ float cstate[64 * 32];   //  8 KB

  const int tid  = threadIdx.x;
  const int lane = tid & 31;
  const int wv   = tid >> 5;               // wave id 0..7 -> n-subtile
  const int dir  = blockIdx.x >> 4;        // 0 fwd, 1 bwd
  const int hcol0 = (blockIdx.x & 15) * 32;

  const unsigned short* wih = dir ? wih_b : wih_f;
  const unsigned short* whh = dir ? whh_b : whh_f;
  const float* bih = dir ? bih_b : bih_f;
  const float* bhh = dir ? bhh_b : bhh_f;
  unsigned short* hseq = dir ? hseq_b : hseq_f;

  // per-thread fixed h-column (phase-3 mapping)
  const int jloc = tid & 31;
  const int hc   = hcol0 + jloc;
  const float bi0 = bih[hc]          + bhh[hc];
  const float bf1 = bih[HH + hc]     + bhh[HH + hc];
  const float bg2 = bih[2 * HH + hc] + bhh[2 * HH + hc];
  const float bo3 = bih[3 * HH + hc] + bhh[3 * HH + hc];
  int lenq[8];                                 // lengths for my 8 cells
  for (int k = 0; k < 8; ++k) lenq[k] = lengths[(tid >> 5) + k * 8];

  for (int q = tid; q < 64 * 32; q += 256) cstate[q] = 0.0f;

  // B-fragment column: lane holds one gate-column, two 8-wide K-runs (k=8h.., 16+8h..)
  const int lane15 = lane & 15;
  const int lcolB = wv * 16 + lane15;                         // local gate col 0..127
  const int gcolB = (lcolB >> 5) * HH + hcol0 + (lcolB & 31); // W row (gate index)
  const int bkoff = (lane >> 4) << 3;                         // 0 or 8
  const int akoff = (lane >> 4) << 3;                         // 0 or 8
  const int lhalf = (lane >> 4) << 3;                         // C-layout row half

  const unsigned short* bx = wih + (size_t)gcolB * DD + bkoff;
  const unsigned short* bh = whh + (size_t)gcolB * HH + bkoff;

  for (int step = 0; step < TT; ++step) {
    const int t = dir ? (TT - 1 - step) : step;
    const unsigned short* hcur = hbuf + ((size_t)dir * 2 + (size_t)(step & 1)) * (BB * HH);
    unsigned short* hnext      = hbuf + ((size_t)dir * 2 + (size_t)((step + 1) & 1)) * (BB * HH);

    // per-m-tile A base pointers for this step
    const unsigned short* ax[4];
    const unsigned short* ah[4];
    for (int mt = 0; mt < 4; ++mt) {
      const int arow = mt * 16 + lane15;      // batch row
      ax[mt] = xbf + ((size_t)arow * TT + t) * DD + akoff;
      ah[mt] = hcur + (size_t)arow * HH + akoff;
    }

    auto load_frags = [&](Frag16& bf_, Frag16* af_,
                          const unsigned short* bp,
                          const unsigned short* const* ap, int koff) {
      bf_.u[0] = *(const uint4*)(bp + koff);
      bf_.u[1] = *(const uint4*)(bp + koff + 16);
#pragma unroll
      for (int mt = 0; mt < 4; ++mt) {
        af_[mt].u[0] = *(const uint4*)(ap[mt] + koff);
        af_[mt].u[1] = *(const uint4*)(ap[mt] + koff + 16);
      }
    };

    // ---- fused input + recurrent GEMM, double-buffered fragments ----
    v8f acc[4];
#pragma unroll
    for (int mt = 0; mt < 4; ++mt)
#pragma unroll
      for (int r = 0; r < 8; ++r) acc[mt][r] = 0.0f;

    Frag16 bfr[2];
    Frag16 afr[2][4];
    load_frags(bfr[0], afr[0], bx, ax, 0);

    // x phase: K = 768 -> 24 chunks of 32
    for (int kc = 0; kc < 24; ++kc) {
      const int cur = kc & 1, nxt = cur ^ 1;
      if (kc < 23) load_frags(bfr[nxt], afr[nxt], bx, ax, (kc + 1) * 32);
      else         load_frags(bfr[nxt], afr[nxt], bh, ah, 0);   // cross-phase prefetch
#pragma unroll
      for (int mt = 0; mt < 4; ++mt)
        acc[mt] = __builtin_amdgcn_wmma_f32_16x16x32_bf16(
            false, afr[cur][mt].v, false, bfr[cur].v, (short)0, acc[mt], false, false);
    }
    // h phase: K = 512 -> 16 chunks of 32 (parity continues: cur = kh&1)
    for (int kh = 0; kh < 16; ++kh) {
      const int cur = kh & 1, nxt = cur ^ 1;
      if (kh < 15) load_frags(bfr[nxt], afr[nxt], bh, ah, (kh + 1) * 32);
#pragma unroll
      for (int mt = 0; mt < 4; ++mt)
        acc[mt] = __builtin_amdgcn_wmma_f32_16x16x32_bf16(
            false, afr[cur][mt].v, false, bfr[cur].v, (short)0, acc[mt], false, false);
    }

    // D layout: lane holds col n = lane&15, rows (lane>=16? 8:0)+r
#pragma unroll
    for (int mt = 0; mt < 4; ++mt)
#pragma unroll
      for (int r = 0; r < 8; ++r)
        gates[(mt * 16 + lhalf + r) * 128 + lcolB] = acc[mt][r];

    // warm L2/WGP$ with next step's X rows while we finish this step
    if (step + 1 < TT) {
      const int tn = dir ? (TT - 2 - step) : (step + 1);
      for (int p = tid; p < 768; p += 256) {          // 64 rows x 12 lines(128B)
        const int row = p / 12, seg = p % 12;
        __builtin_prefetch(xbf + ((size_t)row * TT + tn) * DD + seg * 64, 0, 1);
      }
    }
    __syncthreads();

    // ---- LSTM cell elementwise ----
    for (int k = 0; k < 8; ++k) {
      const int b = (tid >> 5) + k * 8;
      const float* gr = gates + b * 128;
      const float xi = gr[jloc]      + bi0;
      const float xf = gr[32 + jloc] + bf1;
      const float xg = gr[64 + jloc] + bg2;
      const float xo = gr[96 + jloc] + bo3;
      const float i_ = sigf(xi), f_ = sigf(xf), o_ = sigf(xo);
      const float g_ = tanhf(xg);
      const int ci = b * 32 + jloc;
      const float c_old = cstate[ci];
      const float c_new = f_ * c_old + i_ * g_;
      const float h_new = o_ * tanhf(c_new);
      const bool m = (t < lenq[k]);
      cstate[ci] = m ? c_new : c_old;
      const float h_prev = bf2f(hcur[b * HH + hc]);
      hnext[b * HH + hc] = f2bf(m ? h_new : h_prev);
      hseq[((size_t)b * TT + t) * HH + hc] = f2bf(m ? h_new : 0.0f);
    }

    // ---- device-wide step barrier (all 32 WGs) ----
    __threadfence();
    __syncthreads();
    if (tid == 0) {
      asm volatile("s_wait_storecnt 0x0\n\t"
                   "global_wb scope:SCOPE_DEV\n\t"
                   "s_wait_storecnt 0x0" ::: "memory");
      __hip_atomic_fetch_add(cnt, 1u, __ATOMIC_RELEASE, __HIP_MEMORY_SCOPE_AGENT);
      const unsigned target = (unsigned)(NWG_LSTM) * (unsigned)(step + 1);
      while (__hip_atomic_load(cnt, __ATOMIC_ACQUIRE, __HIP_MEMORY_SCOPE_AGENT) < target)
        __builtin_amdgcn_s_sleep(2);
      asm volatile("global_inv scope:SCOPE_DEV\n\t"
                   "s_wait_loadcnt 0x0" ::: "memory");
    }
    __syncthreads();
  }
}

// ---------------- emissions: (B*T,1024) @ W_clf^T + b ----------------
__global__ void emissions_kernel(const unsigned short* __restrict__ hf,
                                 const unsigned short* __restrict__ hb,
                                 const float* __restrict__ wclf,   // (NT, 2H)
                                 const float* __restrict__ bclf,
                                 float* __restrict__ em)           // (B*T, NT)
{
  const int idx = blockIdx.x * 256 + threadIdx.x;   // < B*T*NT
  const int bt = idx >> 4, n = idx & 15;
  const unsigned short* pf = hf + (size_t)bt * HH;
  const unsigned short* pb = hb + (size_t)bt * HH;
  const float* w = wclf + (size_t)n * (2 * HH);
  float acc = bclf[n];
#pragma unroll 4
  for (int j = 0; j < HH; ++j) acc += bf2f(pf[j]) * w[j];
#pragma unroll 4
  for (int j = 0; j < HH; ++j) acc += bf2f(pb[j]) * w[HH + j];
  em[idx] = acc;
}

// ---------------- CRF numerator (gold-path score) ----------------
__global__ void crf_num_kernel(const float* __restrict__ em,
                               const int* __restrict__ targets,
                               const int* __restrict__ lengths,
                               const float* __restrict__ startt,
                               const float* __restrict__ endt,
                               const float* __restrict__ trans,
                               float* __restrict__ numer)
{
  const int b = threadIdx.x;
  if (b >= BB) return;
  const int* tg = targets + (size_t)b * TT;
  const int len = lengths[b];
  int prev = tg[0];
  float score = startt[prev] + em[((size_t)b * TT) * NTAGS + prev];
  for (int t = 1; t < TT; ++t) {
    if (t < len) {
      const int tt2 = tg[t];
      score += trans[prev * NTAGS + tt2] + em[((size_t)b * TT + t) * NTAGS + tt2];
      prev = tt2;
    }
  }
  numer[b] = score + endt[prev];
}

// ---------------- CRF forward (logZ) + llh ----------------
__global__ void crf_den_kernel(const float* __restrict__ em,
                               const int* __restrict__ lengths,
                               const float* __restrict__ startt,
                               const float* __restrict__ endt,
                               const float* __restrict__ trans,
                               const float* __restrict__ numer,
                               float* __restrict__ llh)
{
  const int b = blockIdx.x;
  const int n = threadIdx.x;
  __shared__ float alpha[NTAGS];
  __shared__ float tr[NTAGS * NTAGS];
  for (int i = n; i < NTAGS * NTAGS; i += 32) tr[i] = trans[i];
  if (n < NTAGS) alpha[n] = startt[n] + em[((size_t)b * TT) * NTAGS + n];
  __syncthreads();
  const int len = lengths[b];
  for (int t = 1; t < TT; ++t) {
    float nv = 0.0f;
    if (n < NTAGS) {
      float mx = -3.4e38f;
      for (int m = 0; m < NTAGS; ++m) mx = fmaxf(mx, alpha[m] + tr[m * NTAGS + n]);
      float s = 0.0f;
      for (int m = 0; m < NTAGS; ++m) s += __expf(alpha[m] + tr[m * NTAGS + n] - mx);
      nv = mx + __logf(s) + em[((size_t)b * TT + t) * NTAGS + n];
    }
    __syncthreads();
    if (n < NTAGS && t < len) alpha[n] = nv;
    __syncthreads();
  }
  if (n == 0) {
    float mx = -3.4e38f;
    for (int i = 0; i < NTAGS; ++i) mx = fmaxf(mx, alpha[i] + endt[i]);
    float s = 0.0f;
    for (int i = 0; i < NTAGS; ++i) s += __expf(alpha[i] + endt[i] - mx);
    llh[b] = numer[b] - (mx + __logf(s));
  }
}

// ---------------- Viterbi (fwd + backtrack) ----------------
__global__ void viterbi_kernel(const float* __restrict__ em,
                               const int* __restrict__ lengths,
                               const float* __restrict__ startt,
                               const float* __restrict__ endt,
                               const float* __restrict__ trans,
                               unsigned char* __restrict__ bp,   // (B, T-1, NT)
                               float* __restrict__ out)          // d_out
{
  const int b = blockIdx.x;
  const int n = threadIdx.x;
  __shared__ float sc[NTAGS];
  __shared__ float tr[NTAGS * NTAGS];
  for (int i = n; i < NTAGS * NTAGS; i += 32) tr[i] = trans[i];
  if (n < NTAGS) sc[n] = startt[n] + em[((size_t)b * TT) * NTAGS + n];
  __syncthreads();
  const int len = lengths[b];
  for (int t = 1; t < TT; ++t) {
    float best = -3.4e38f;
    int arg = 0;
    if (n < NTAGS) {
      for (int m = 0; m < NTAGS; ++m) {
        const float v = sc[m] + tr[m * NTAGS + n];
        if (v > best) { best = v; arg = m; }
      }
    }
    const float nv = best + em[((size_t)b * TT + t) * NTAGS + n];
    __syncthreads();
    if (n < NTAGS) {
      if (t < len) sc[n] = nv;
      bp[((size_t)b * (TT - 1) + (t - 1)) * NTAGS + n] = (unsigned char)arg;
    }
    __syncthreads();
  }
  if (n == 0) {
    float best = -3.4e38f;
    int cur = 0;
    for (int i = 0; i < NTAGS; ++i) {
      const float v = sc[i] + endt[i];
      if (v > best) { best = v; cur = i; }
    }
    for (int t = TT - 1; t >= 1; --t) {
      out[1 + (size_t)b * TT + t] = (t < len) ? (float)cur : 0.0f;
      if (t < len) cur = bp[((size_t)b * (TT - 1) + (t - 1)) * NTAGS + cur];
    }
    out[1 + (size_t)b * TT + 0] = (float)cur;   // t=0 always unmasked (len >= T/2)
  }
}

__global__ void finalize_kernel(const float* __restrict__ llh, float* __restrict__ out) {
  if (threadIdx.x == 0) {
    float s = 0.0f;
    for (int b = 0; b < BB; ++b) s += llh[b];
    out[0] = -s / (float)BB;
  }
}

// ---------------- host launcher ----------------
extern "C" void kernel_launch(void* const* d_in, const int* in_sizes, int n_in,
                              void* d_out, int out_size, void* d_ws, size_t ws_size,
                              hipStream_t stream) {
  (void)in_sizes; (void)n_in; (void)out_size; (void)ws_size;
  const float* x      = (const float*)d_in[0];
  const int*   lens   = (const int*)d_in[2];
  const int*   tgts   = (const int*)d_in[3];
  const float* Wih_f  = (const float*)d_in[4];
  const float* Whh_f  = (const float*)d_in[5];
  const float* bih_f  = (const float*)d_in[6];
  const float* bhh_f  = (const float*)d_in[7];
  const float* Wih_b  = (const float*)d_in[8];
  const float* Whh_b  = (const float*)d_in[9];
  const float* bih_b  = (const float*)d_in[10];
  const float* bhh_b  = (const float*)d_in[11];
  const float* Wclf   = (const float*)d_in[12];
  const float* bclf   = (const float*)d_in[13];
  const float* startt = (const float*)d_in[14];
  const float* endt   = (const float*)d_in[15];
  const float* trans  = (const float*)d_in[16];
  float* out = (float*)d_out;

  char* ws = (char*)d_ws;
  size_t off = 0;
  auto take = [&](size_t bytes) -> char* {
    char* p = ws + off;
    off = (off + bytes + 255) & ~(size_t)255;
    return p;
  };
  unsigned short* XBF   = (unsigned short*)take((size_t)BB * TT * DD * 2);
  unsigned short* WIHF  = (unsigned short*)take((size_t)GG * DD * 2);
  unsigned short* WHHF  = (unsigned short*)take((size_t)GG * HH * 2);
  unsigned short* WIHB  = (unsigned short*)take((size_t)GG * DD * 2);
  unsigned short* WHHB  = (unsigned short*)take((size_t)GG * HH * 2);
  unsigned short* HBUF  = (unsigned short*)take((size_t)4 * BB * HH * 2);
  unsigned short* HSEQF = (unsigned short*)take((size_t)BB * TT * HH * 2);
  unsigned short* HSEQB = (unsigned short*)take((size_t)BB * TT * HH * 2);
  float*          EM    = (float*)take((size_t)BB * TT * NTAGS * 4);
  unsigned char*  BP    = (unsigned char*)take((size_t)BB * (TT - 1) * NTAGS);
  float*          NUMER = (float*)take(BB * 4);
  float*          LLH   = (float*)take(BB * 4);
  unsigned int*   CNT   = (unsigned int*)take(256);

  // conversions + state init
  cvt_f32_bf16<<<2048, 256, 0, stream>>>(x, XBF, BB * TT * DD);
  cvt_f32_bf16<<<512, 256, 0, stream>>>(Wih_f, WIHF, GG * DD);
  cvt_f32_bf16<<<512, 256, 0, stream>>>(Whh_f, WHHF, GG * HH);
  cvt_f32_bf16<<<512, 256, 0, stream>>>(Wih_b, WIHB, GG * DD);
  cvt_f32_bf16<<<512, 256, 0, stream>>>(Whh_b, WHHB, GG * HH);
  zero_u16<<<64, 256, 0, stream>>>(HBUF, 4 * BB * HH);
  zero_u32<<<1, 32, 0, stream>>>(CNT, 1);

  // persistent bidirectional LSTM recurrence (WMMA)
  lstm_kernel<<<NWG_LSTM, 256, 0, stream>>>(XBF, WIHF, WHHF, WIHB, WHHB,
                                            bih_f, bhh_f, bih_b, bhh_b,
                                            lens, HBUF, HSEQF, HSEQB, CNT);

  // emissions + CRF + viterbi
  emissions_kernel<<<(BB * TT * NTAGS) / 256, 256, 0, stream>>>(HSEQF, HSEQB, Wclf, bclf, EM);
  crf_num_kernel<<<1, 64, 0, stream>>>(EM, tgts, lens, startt, endt, trans, NUMER);
  crf_den_kernel<<<BB, 32, 0, stream>>>(EM, lens, startt, endt, trans, NUMER, LLH);
  viterbi_kernel<<<BB, 32, 0, stream>>>(EM, lens, startt, endt, trans, BP, out);
  finalize_kernel<<<1, 32, 0, stream>>>(LLH, out);
}
